// KANMultiheadAttention_62989990363549
// MI455X (gfx1250) — compile-verified, matrix operand order checked by reading, not run on hardware
//
#include <hip/hip_runtime.h>
#include <hip/hip_bf16.h>
#include <math.h>

// ---------------- Types for CDNA5 WMMA ----------------
typedef __attribute__((ext_vector_type(16))) __bf16 v16bf;
typedef __attribute__((ext_vector_type(8)))  float  v8f;
typedef __attribute__((ext_vector_type(4)))  unsigned int u32x4;
typedef __attribute__((ext_vector_type(8)))  int          i32x8;
typedef __attribute__((ext_vector_type(4)))  int          i32x4;

union FragBF {            // 16 bf16 = 32 bytes = one WMMA A/B fragment per lane
    v16bf          v;
    uint4          q[2];
    unsigned short h[16];
};

__device__ __forceinline__ unsigned short f2bf(float f) {
    union { float f; unsigned int u; } c; c.f = f;
    unsigned int u = c.u;
    u += 0x7FFFu + ((u >> 16) & 1u);      // round-to-nearest-even
    return (unsigned short)(u >> 16);
}

// Problem constants
#define BB   2
#define LL   2048
#define EE   1024
#define HH   16
#define DD   64
#define NB   7             // GRID_SIZE + SPLINE_ORDER
#define NROW (BB*LL)       // 4096
#define KTOT (EE + EE*NB)  // 8192
#define NBH  (BB*HH)       // 32

// ---------------------------------------------------------------------------
// TDM: issue a 2D tensor_load_to_lds for a [128 rows x 32 cols] bf16 tile out
// of a row-major [rows x KTOT] bf16 matrix. Tracked by TENSORcnt.
// D# group0: count=1 | lds_addr | global_addr[56:0] | type=2
// D# group1: data_size=1(2B), tensor_dim0=KTOT, tensor_dim1=rows,
//            tile_dim0=32, tile_dim1=128, tensor_dim0_stride=KTOT
// ---------------------------------------------------------------------------
__device__ __forceinline__ unsigned int lds_off(const void* p) {
    // low 32 bits of the generic address of an LDS pointer == LDS byte offset
    return (unsigned int)(size_t)p;
}

__device__ __forceinline__ void tdm_load_tile(const unsigned short* gptr,
                                              unsigned int ldsoff,
                                              unsigned int rows) {
    unsigned long long ga = (unsigned long long)(size_t)gptr;
    u32x4 g0;
    g0[0] = 1u;                                             // count=1, user mode
    g0[1] = ldsoff;                                         // lds_addr (bytes)
    g0[2] = (unsigned int)ga;                               // global_addr[31:0]
    g0[3] = ((unsigned int)(ga >> 32) & 0x01FFFFFFu) | 0x80000000u; // [56:32] | type=2
    i32x8 g1;
    g1[0] = 0x00010000;                                     // wg_mask=0, data_size=1 (2B)
    g1[1] = (int)(((unsigned)KTOT & 0xFFFFu) << 16);        // tensor_dim0 lo16 @ [63:48]
    g1[2] = (int)(((unsigned)KTOT >> 16) | ((rows & 0xFFFFu) << 16)); // td0 hi | td1 lo
    g1[3] = (int)((rows >> 16) | (32u << 16));              // td1 hi | tile_dim0=32
    g1[4] = 128;                                            // tile_dim1=128, tile_dim2=0
    g1[5] = KTOT;                                           // tensor_dim0_stride lo32
    g1[6] = 0;                                              // td0_stride hi | td1_stride lo
    g1[7] = 0;
    i32x4 z4 = {0, 0, 0, 0};
#if __clang_major__ >= 23
    i32x8 z8 = {0, 0, 0, 0, 0, 0, 0, 0};
    __builtin_amdgcn_tensor_load_to_lds(g0, g1, z4, z4, z8, 0);
#else
    __builtin_amdgcn_tensor_load_to_lds(g0, g1, z4, z4, 0);
#endif
}

// ---------------------------------------------------------------------------
// Kernel 1: pack weights: W[o][k] bf16, k<1024 -> base_w[o][k],
//           else spline_w[o][f][s]*scaler[o][f], f=(k-1024)/7, s=(k-1024)%7
// ---------------------------------------------------------------------------
__global__ __launch_bounds__(256)
void kan_wprep(const float* __restrict__ base_w,
               const float* __restrict__ spline_w,
               const float* __restrict__ scaler,
               unsigned short* __restrict__ W) {
    int idx = blockIdx.x * 256 + threadIdx.x;      // 1024*8192
    int o = idx >> 13;
    int k = idx & (KTOT - 1);
    float w;
    if (k < EE) {
        w = base_w[o * EE + k];
    } else {
        int kk = k - EE;
        int f = kk / NB;
        int s = kk - f * NB;
        w = spline_w[(o * EE + f) * NB + s] * scaler[o * EE + f];
    }
    W[(size_t)o * KTOT + k] = f2bf(w);
}

// ---------------------------------------------------------------------------
// Kernel 2: build A = [x | b_splines(x)] bf16, shape [4096, 8192]
// Uniform grid: g_t = -2.5 + 0.5*t, t = 0..10 (GRID_SIZE=4, k=3)
// ---------------------------------------------------------------------------
__global__ __launch_bounds__(256)
void kan_sb(const float* __restrict__ X, unsigned short* __restrict__ A) {
    int idx = blockIdx.x * 256 + threadIdx.x;      // 4096*1024
    int row = idx >> 10;
    int f   = idx & (EE - 1);
    float x = X[idx];

    float cur[10];
    #pragma unroll
    for (int i = 0; i < 10; ++i) {
        float g0 = -2.5f + 0.5f * i;
        float g1 = g0 + 0.5f;
        cur[i] = (x >= g0 && x < g1) ? 1.0f : 0.0f;
    }
    #pragma unroll
    for (int p = 1; p <= 3; ++p) {
        float inv = 1.0f / (0.5f * p);
        #pragma unroll
        for (int i = 0; i < 9; ++i) {
            if (i < 10 - p) {
                float gi   = -2.5f + 0.5f * i;
                float gip1 = gi + 0.5f * (p + 1);
                float left  = (x - gi) * inv;
                float right = (gip1 - x) * inv;
                cur[i] = left * cur[i] + right * cur[i + 1];
            }
        }
    }

    unsigned short* dst = A + (size_t)row * KTOT;
    dst[f] = f2bf(x);
    #pragma unroll
    for (int s = 0; s < NB; ++s)
        dst[EE + f * NB + s] = f2bf(cur[s]);
}

// ---------------------------------------------------------------------------
// Kernel 3: KAN GEMM  Y[4096,1024] = A[4096,8192] * W[1024,8192]^T (bf16->f32)
// 128x128 block tile, 8 waves x (64x32), TDM double-buffered LDS staging.
// mode 0: store bf16 [B,H,L,D]; mode 1: store bf16 [B,H,D,L] (V transposed)
// ---------------------------------------------------------------------------
__global__ __launch_bounds__(256)
void kan_gemm(const unsigned short* __restrict__ A,
              const unsigned short* __restrict__ W,
              unsigned short* __restrict__ Y, int mode) {
    __shared__ unsigned short sA[2][128 * 32];
    __shared__ unsigned short sB[2][128 * 32];

    int wave = threadIdx.x >> 5;
    int lane = threadIdx.x & 31;
    int half = lane >> 4;
    int mn   = lane & 15;
    int bm   = blockIdx.x >> 3;           // 0..31 (M block)
    int bn   = blockIdx.x & 7;            // 0..7  (N block)
    int wm   = wave >> 2;                 // 0..1
    int wn   = wave & 3;                  // 0..3

    const unsigned short* Abase = A + (size_t)bm * 128 * KTOT;
    const unsigned short* Wbase = W + (size_t)bn * 128 * KTOT;

    if (wave == 0) {                      // preload buffer 0 (K-chunk 0)
        tdm_load_tile(Abase, lds_off(&sA[0][0]), NROW);
        tdm_load_tile(Wbase, lds_off(&sB[0][0]), EE);
    }

    v8f acc[4][2] = {};
    const int nIter = KTOT / 32;          // 256
    for (int it = 0; it < nIter; ++it) {
        int cur = it & 1;
        if (wave == 0) {
            if (it + 1 < nIter) {         // prefetch next chunk into other buffer
                int kc = (it + 1) * 32;
                tdm_load_tile(Abase + kc, lds_off(&sA[cur ^ 1][0]), NROW);
                tdm_load_tile(Wbase + kc, lds_off(&sB[cur ^ 1][0]), EE);
                __builtin_amdgcn_s_wait_tensorcnt(2);   // current pair complete
            } else {
                __builtin_amdgcn_s_wait_tensorcnt(0);
            }
        }
        __syncthreads();                  // LDS tile 'cur' is ready for all waves

        FragBF bfr[2];
        #pragma unroll
        for (int nt = 0; nt < 2; ++nt) {
            const uint4* p = (const uint4*)&sB[cur][(wn * 32 + nt * 16 + mn) * 32];
            bfr[nt].q[0] = p[half];
            bfr[nt].q[1] = p[half + 2];
        }
        #pragma unroll
        for (int mt = 0; mt < 4; ++mt) {
            FragBF a;
            const uint4* p = (const uint4*)&sA[cur][(wm * 64 + mt * 16 + mn) * 32];
            a.q[0] = p[half];
            a.q[1] = p[half + 2];
            #pragma unroll
            for (int nt = 0; nt < 2; ++nt)
                acc[mt][nt] = __builtin_amdgcn_wmma_f32_16x16x32_bf16(
                    false, a.v, false, bfr[nt].v, (short)0, acc[mt][nt], false, false);
        }
        __syncthreads();                  // done reading 'cur' before it is refilled
    }

    int colbase = bn * 128 + wn * 32;
    int rowbase = bm * 128 + wm * 64;
    #pragma unroll
    for (int mt = 0; mt < 4; ++mt) {
        #pragma unroll
        for (int nt = 0; nt < 2; ++nt) {
            int col = colbase + nt * 16 + mn;
            int h   = col >> 6;
            int dd  = col & (DD - 1);
            #pragma unroll
            for (int r = 0; r < 8; ++r) {
                int row = rowbase + mt * 16 + r + half * 8;
                int bb  = row >> 11;
                int l   = row & (LL - 1);
                unsigned short val = f2bf(acc[mt][nt][r]);
                if (mode == 0)
                    Y[(((size_t)bb * HH + h) * LL + l) * DD + dd] = val;
                else
                    Y[(((size_t)bb * HH + h) * DD + dd) * LL + l] = val;
            }
        }
    }
}

// ---------------------------------------------------------------------------
// Kernel 4: scores[bh, q, k] = (Q . K^T) / 8   (bf16 WMMA, K=64)
// ---------------------------------------------------------------------------
__global__ __launch_bounds__(256)
void attn_scores(const unsigned short* __restrict__ Qb,
                 const unsigned short* __restrict__ Kb,
                 float* __restrict__ S) {
    int wave = threadIdx.x >> 5;
    int lane = threadIdx.x & 31;
    int t    = blockIdx.x * 8 + wave;     // 32 * 128 * 128 tiles
    int bh   = t >> 14;
    int rem  = t & 16383;
    int tq   = rem >> 7;
    int tk   = rem & 127;
    int half = lane >> 4;
    int mn   = lane & 15;

    const uint4* Ab = (const uint4*)(Qb + (size_t)(bh * LL + tq * 16 + mn) * DD);
    const uint4* Bb = (const uint4*)(Kb + (size_t)(bh * LL + tk * 16 + mn) * DD);

    v8f acc = {};
    #pragma unroll
    for (int kc = 0; kc < DD; kc += 32) {
        FragBF a, b;
        int qi = (kc >> 3) + half;
        a.q[0] = Ab[qi];  a.q[1] = Ab[qi + 2];
        b.q[0] = Bb[qi];  b.q[1] = Bb[qi + 2];
        acc = __builtin_amdgcn_wmma_f32_16x16x32_bf16(
                  false, a.v, false, b.v, (short)0, acc, false, false);
    }
    #pragma unroll
    for (int r = 0; r < 8; ++r) {
        int qrow = tq * 16 + r + half * 8;
        S[((size_t)bh * LL + qrow) * LL + tk * 16 + mn] = acc[r] * 0.125f;
    }
}

// ---------------------------------------------------------------------------
// Kernel 5: per-row max and 1/sum(exp) over 2048 columns
// ---------------------------------------------------------------------------
__global__ __launch_bounds__(256)
void row_stats(const float* __restrict__ S,
               float* __restrict__ mx, float* __restrict__ inv) {
    __shared__ float red[256];
    size_t row = blockIdx.x;
    const float* p = S + row * LL;
    int tid = threadIdx.x;

    float m = -INFINITY;
    for (int i = tid; i < LL; i += 256) m = fmaxf(m, p[i]);
    red[tid] = m; __syncthreads();
    for (int s = 128; s > 0; s >>= 1) {
        if (tid < s) red[tid] = fmaxf(red[tid], red[tid + s]);
        __syncthreads();
    }
    m = red[0]; __syncthreads();

    float sum = 0.0f;
    for (int i = tid; i < LL; i += 256) sum += __expf(p[i] - m);
    red[tid] = sum; __syncthreads();
    for (int s = 128; s > 0; s >>= 1) {
        if (tid < s) red[tid] += red[tid + s];
        __syncthreads();
    }
    if (tid == 0) { mx[row] = m; inv[row] = 1.0f / red[0]; }
}

// ---------------------------------------------------------------------------
// Kernel 6: in-place softmax normalize (float4 per thread)
// ---------------------------------------------------------------------------
__global__ __launch_bounds__(256)
void norm_scores(float* __restrict__ S,
                 const float* __restrict__ mx, const float* __restrict__ inv) {
    size_t e = ((size_t)blockIdx.x * 256 + threadIdx.x) * 4;
    size_t row = e >> 11;
    float m = mx[row], iv = inv[row];
    float4 v = *(const float4*)(S + e);
    v.x = __expf(v.x - m) * iv;
    v.y = __expf(v.y - m) * iv;
    v.z = __expf(v.z - m) * iv;
    v.w = __expf(v.w - m) * iv;
    *(float4*)(S + e) = v;
}

// ---------------------------------------------------------------------------
// Kernel 7: out[bh, l, d] = W[2048,2048] * V[2048,64]   (weights f32 -> bf16,
// V pre-transposed [bh, 64, 2048] bf16), write attn_output f32 [B,L,E]
// ---------------------------------------------------------------------------
__global__ __launch_bounds__(256)
void attn_av(const float* __restrict__ Wn,
             const unsigned short* __restrict__ Vt,
             float* __restrict__ Out) {
    int wave = threadIdx.x >> 5;
    int lane = threadIdx.x & 31;
    int t    = blockIdx.x * 8 + wave;     // 32 * 128 * 4 tiles
    int bh   = t >> 9;
    int rem  = t & 511;
    int tl   = rem >> 2;
    int td   = rem & 3;
    int half = lane >> 4;
    int mn   = lane & 15;

    const float4* Ar = (const float4*)(Wn + (size_t)(bh * LL + tl * 16 + mn) * LL);
    const uint4*  Bb = (const uint4*)(Vt + (size_t)(bh * DD + td * 16 + mn) * LL);

    v8f acc = {};
    for (int kc = 0; kc < LL; kc += 32) {
        FragBF a, b;
        int f4 = (kc >> 2) + half * 2;
        float4 x0 = Ar[f4], x1 = Ar[f4 + 1], x2 = Ar[f4 + 4], x3 = Ar[f4 + 5];
        a.h[0]  = f2bf(x0.x); a.h[1]  = f2bf(x0.y); a.h[2]  = f2bf(x0.z); a.h[3]  = f2bf(x0.w);
        a.h[4]  = f2bf(x1.x); a.h[5]  = f2bf(x1.y); a.h[6]  = f2bf(x1.z); a.h[7]  = f2bf(x1.w);
        a.h[8]  = f2bf(x2.x); a.h[9]  = f2bf(x2.y); a.h[10] = f2bf(x2.z); a.h[11] = f2bf(x2.w);
        a.h[12] = f2bf(x3.x); a.h[13] = f2bf(x3.y); a.h[14] = f2bf(x3.z); a.h[15] = f2bf(x3.w);
        int qi = (kc >> 3) + half;
        b.q[0] = Bb[qi];  b.q[1] = Bb[qi + 2];
        acc = __builtin_amdgcn_wmma_f32_16x16x32_bf16(
                  false, a.v, false, b.v, (short)0, acc, false, false);
    }

    int b  = bh >> 4;
    int h  = bh & (HH - 1);
    int d  = td * 16 + mn;
    #pragma unroll
    for (int r = 0; r < 8; ++r) {
        int l = tl * 16 + r + half * 8;
        Out[((size_t)b * LL + l) * EE + h * DD + d] = acc[r];
    }
}

// ---------------------------------------------------------------------------
extern "C" void kernel_launch(void* const* d_in, const int* in_sizes, int n_in,
                              void* d_out, int out_size, void* d_ws, size_t ws_size,
                              hipStream_t stream) {
    (void)in_sizes; (void)n_in; (void)out_size; (void)ws_size;

    // Workspace layout (bytes)
    char* ws = (char*)d_ws;
    unsigned short* A_buf = (unsigned short*)(ws);                     // 64 MB
    unsigned short* W_buf = (unsigned short*)(ws + 67108864);          // 16 MB
    unsigned short* Qb    = (unsigned short*)(ws + 83886080);          // 8 MB
    unsigned short* Kb    = (unsigned short*)(ws + 92274688);          // 8 MB
    unsigned short* Vt    = (unsigned short*)(ws + 100663296);         // 8 MB
    float*          mx    = (float*)(ws + 109051904);                  // 256 KB
    float*          inv   = (float*)(ws + 109314048);                  // 256 KB

    float* out    = (float*)d_out;                 // attn_output [B,L,E]
    float* scores = out + (size_t)BB * LL * EE;    // attn_weights [B,H,L,L]

    // --- KAN projections: q, k, v (sequential; A_buf/W_buf reused) ---
    for (int p = 0; p < 3; ++p) {
        const float* base_w   = (const float*)d_in[3 + 3 * p];
        const float* spline_w = (const float*)d_in[4 + 3 * p];
        const float* scaler   = (const float*)d_in[5 + 3 * p];
        const float* X        = (const float*)d_in[p];

        kan_wprep<<<(EE * KTOT) / 256, 256, 0, stream>>>(base_w, spline_w, scaler, W_buf);
        kan_sb   <<<(NROW * EE) / 256, 256, 0, stream>>>(X, A_buf);
        unsigned short* Y = (p == 0) ? Qb : (p == 1) ? Kb : Vt;
        kan_gemm <<<(NROW / 128) * (EE / 128), 256, 0, stream>>>(A_buf, W_buf, Y,
                                                                 (p == 2) ? 1 : 0);
    }

    // --- Attention ---
    attn_scores<<<NBH * (LL / 16) * (LL / 16) / 8, 256, 0, stream>>>(Qb, Kb, scores);
    row_stats  <<<NBH * LL, 256, 0, stream>>>(scores, mx, inv);
    norm_scores<<<((size_t)NBH * LL * LL) / 1024, 256, 0, stream>>>(scores, mx, inv);
    attn_av    <<<NBH * (LL / 16) * (DD / 16) / 8, 256, 0, stream>>>(scores, Vt, out);
}
